// vanillagnn_70093866270945
// MI455X (gfx1250) — compile-verified, matrix operand order checked by reading, not run on hardware
//
#include <hip/hip_runtime.h>
#include <hip/hip_bf16.h>

#define NN 50000
#define NE 800000
#define NG 64
#define FN 8
#define FE 4
#define DD 16
#define DE 8
#define EPSV 1e-5f

typedef _Float16 v16h __attribute__((ext_vector_type(16)));
typedef _Float16 v8h  __attribute__((ext_vector_type(8)));
typedef float    v8f  __attribute__((ext_vector_type(8)));

// CDNA5 has native V_TANH_F32 (listed as a TRANS op in the ISA's S_DELAY_ALU
// section); use the builtin when the toolchain exposes it, else a fast
// exp+rcp expansion (v_exp_f32 + v_rcp_f32, no IEEE divide).
__device__ __forceinline__ float fast_tanh(float x) {
#if __has_builtin(__builtin_amdgcn_tanhf)
    return __builtin_amdgcn_tanhf(x);
#else
    x = fminf(fmaxf(x, -15.f), 15.f);
    float e = __expf(2.f * x);
#if __has_builtin(__builtin_amdgcn_rcpf)
    return (e - 1.f) * __builtin_amdgcn_rcpf(e + 1.f);
#else
    return (e - 1.f) / (e + 1.f);
#endif
#endif
}
__device__ __forceinline__ unsigned fkey(float x) {
    unsigned b = __float_as_uint(x);
    return (b & 0x80000000u) ? ~b : (b | 0x80000000u);
}
__device__ __forceinline__ float fkey_inv(unsigned k) {
    unsigned b = (k & 0x80000000u) ? (k & 0x7FFFFFFFu) : ~k;
    return __uint_as_float(b);
}

// ---------------- init scratch (stats + pooling accumulators) ----------------
__global__ void k_init(float* stats, float* psum, float* pcnt, unsigned* pmax) {
    int i = blockIdx.x * blockDim.x + threadIdx.x;
    if (i < 96)       stats[i] = 0.f;
    if (i < NG * DD)  psum[i]  = 0.f;
    if (i < NG)       pcnt[i]  = 0.f;
    if (i < NG * DD)  pmax[i]  = 0x007FFFFFu;  // fkey(-inf)
}

// ---------------- node embed: relu(x@W+b), accumulate BN stats ---------------
__global__ void k_node_stats(const float* __restrict__ x, const float* __restrict__ W,
                             const float* __restrict__ b, float* __restrict__ gsum,
                             float* __restrict__ gsq) {
    __shared__ float ssum[DD], ssq[DD];
    int tid = threadIdx.x;
    if (tid < DD) { ssum[tid] = 0.f; ssq[tid] = 0.f; }
    __syncthreads();
    int n = blockIdx.x * blockDim.x + tid;
    if (n < NN) {
        float xr[FN];
        #pragma unroll
        for (int j = 0; j < FN; ++j) xr[j] = x[n * FN + j];
        #pragma unroll
        for (int c = 0; c < DD; ++c) {
            float y = b[c];
            #pragma unroll
            for (int j = 0; j < FN; ++j) y = fmaf(xr[j], W[j * DD + c], y);
            y = fmaxf(y, 0.f);
            atomicAdd(&ssum[c], y);
            atomicAdd(&ssq[c], y * y);
        }
    }
    __syncthreads();
    if (tid < DD) { atomicAdd(&gsum[tid], ssum[tid]); atomicAdd(&gsq[tid], ssq[tid]); }
}

// ---------------- edge embed: celu(ea@W+b), accumulate BN stats --------------
__global__ void k_edge_stats(const float* __restrict__ ea, const float* __restrict__ W,
                             const float* __restrict__ b, float* __restrict__ gsum,
                             float* __restrict__ gsq) {
    __shared__ float ssum[DE], ssq[DE];
    int tid = threadIdx.x;
    if (tid < DE) { ssum[tid] = 0.f; ssq[tid] = 0.f; }
    __syncthreads();
    int e = blockIdx.x * blockDim.x + tid;
    if (e < NE) {
        float4 v = *reinterpret_cast<const float4*>(ea + (size_t)e * FE);
        #pragma unroll
        for (int k = 0; k < DE; ++k) {
            float z = b[k];
            z = fmaf(v.x, W[0 * DE + k], z);
            z = fmaf(v.y, W[1 * DE + k], z);
            z = fmaf(v.z, W[2 * DE + k], z);
            z = fmaf(v.w, W[3 * DE + k], z);
            z = (z > 0.f) ? z : (__expf(z) - 1.f);
            atomicAdd(&ssum[k], z);
            atomicAdd(&ssq[k], z * z);
        }
    }
    __syncthreads();
    if (tid < DE) { atomicAdd(&gsum[tid], ssum[tid]); atomicAdd(&gsq[tid], ssq[tid]); }
}

// ---------------- fold BN stats into scale/shift -----------------------------
// stats layout: [0]nsum16 [16]nsq16 [32]nscale16 [48]nshift16
//               [64]esum8 [72]esq8 [80]escale8 [88]eshift8
__global__ void k_finalize(const float* gn, const float* ben, const float* ge,
                           const float* bee, float* stats) {
    int t = threadIdx.x;
    if (t < DD) {
        float m = stats[t] / (float)NN;
        float v = stats[16 + t] / (float)NN - m * m;
        float sc = gn[t] * rsqrtf(v + EPSV);
        stats[32 + t] = sc;
        stats[48 + t] = ben[t] - m * sc;
    } else if (t < DD + DE) {
        int c = t - DD;
        float m = stats[64 + c] / (float)NE;
        float v = stats[72 + c] / (float)NE - m * m;
        float sc = ge[c] * rsqrtf(v + EPSV);
        stats[80 + c] = sc;
        stats[88 + c] = bee[c] - m * sc;
    }
}

// ---------------- apply node embed+BN -> h0 ----------------------------------
__global__ void k_node_apply(const float* __restrict__ x, const float* __restrict__ W,
                             const float* __restrict__ b, const float* __restrict__ stats,
                             float* __restrict__ h0) {
    int n = blockIdx.x * blockDim.x + threadIdx.x;
    if (n >= NN) return;
    float xr[FN];
    #pragma unroll
    for (int j = 0; j < FN; ++j) xr[j] = x[n * FN + j];
    #pragma unroll
    for (int c = 0; c < DD; ++c) {
        float y = b[c];
        #pragma unroll
        for (int j = 0; j < FN; ++j) y = fmaf(xr[j], W[j * DD + c], y);
        y = fmaxf(y, 0.f);
        h0[(size_t)n * DD + c] = fmaf(y, stats[32 + c], stats[48 + c]);
    }
}

// ---------------- root term: h_out = bias + h_in @ root ----------------------
__global__ void k_root(const float* __restrict__ h_in, float* __restrict__ h_out,
                       const float* __restrict__ root, const float* __restrict__ bias) {
    int n = blockIdx.x * blockDim.x + threadIdx.x;
    if (n >= NN) return;
    float hr[DD];
    #pragma unroll
    for (int i = 0; i < DD; ++i) hr[i] = h_in[(size_t)n * DD + i];
    #pragma unroll
    for (int o = 0; o < DD; ++o) {
        float acc = bias[o];
        #pragma unroll
        for (int i = 0; i < DD; ++i) acc = fmaf(hr[i], root[i * DD + o], acc);
        h_out[(size_t)n * DD + o] = acc;
    }
}

// ---------------- NNConv message pass: WMMA edge-MLP + fused einsum ----------
// One wave = 16 edges. A = e_block [16 x 8 (K padded to 32)] f16,
// B = W_en tile [8 x 16] f16, one v_wmma_f32_16x16x32_f16 per output tile t.
// Tile t holds w[edge, i=t, o=0..15]; fused: macc += h_src[edge,t] * tanh(wmma).
__global__ __launch_bounds__(256) void k_nnconv(
    const float* __restrict__ h_in, float* __restrict__ h_out,
    const float* __restrict__ edge_attr, const int* __restrict__ eidx,
    const float* __restrict__ W_ee, const float* __restrict__ b_ee,
    const float* __restrict__ W_en, const float* __restrict__ b_en,
    const float* __restrict__ stats) {
    __shared__ __align__(16) _Float16 Wl[DD * DD * DE];  // [(col)*8+k]
    __shared__ float benl[DD * DD];
    __shared__ float hs[8][16 * 17];

    int tid = threadIdx.x;
    {   // stage W_en (f16, column-contiguous in k) and b_en into LDS
        int c = tid;  // 256 threads, 256 columns
        #pragma unroll
        for (int k = 0; k < DE; ++k) Wl[c * DE + k] = (_Float16)W_en[k * (DD * DD) + c];
        benl[c] = b_en[c];
    }
    int wave = tid >> 5;
    int lane = tid & 31;
    int base = (blockIdx.x * 8 + wave) * 16;  // 16 edges per wave; grid exact

    // A matrix: lanes 0-15 hold edge row M=lane, K=0..7 real (rest zero)
    int src_i = 0, dst_i = 0;
    v16h a;
    #pragma unroll
    for (int i = 0; i < 16; ++i) a[i] = (_Float16)0.f;
    if (lane < 16) {
        int e = base + lane;
        src_i = eidx[e];
        dst_i = eidx[NE + e];
        float4 v = *reinterpret_cast<const float4*>(edge_attr + (size_t)e * FE);
        #pragma unroll
        for (int k = 0; k < DE; ++k) {  // recompute edge embed + BN on the fly
            float z = b_ee[k];
            z = fmaf(v.x, W_ee[0 * DE + k], z);
            z = fmaf(v.y, W_ee[1 * DE + k], z);
            z = fmaf(v.z, W_ee[2 * DE + k], z);
            z = fmaf(v.w, W_ee[3 * DE + k], z);
            z = (z > 0.f) ? z : (__expf(z) - 1.f);
            z = fmaf(z, stats[80 + k], stats[88 + k]);
            a[k] = (_Float16)z;
        }
    }
    {   // stage h_in rows of the 16 source nodes: lane -> (edge lane&15, half lane>>4)
        int m = lane & 15, half = lane >> 4;
        int sm = __shfl(src_i, m, 32);
        const float4* hp = reinterpret_cast<const float4*>(h_in + (size_t)sm * DD + half * 8);
        float4 p0 = hp[0], p1 = hp[1];
        float* dl = &hs[wave][m * 17 + half * 8];
        dl[0] = p0.x; dl[1] = p0.y; dl[2] = p0.z; dl[3] = p0.w;
        dl[4] = p1.x; dl[5] = p1.y; dl[6] = p1.z; dl[7] = p1.w;
    }
    int mrow = (lane >> 4) * 8;  // C/D layout: lane handles rows mrow..mrow+7
    int dstv[8];
    #pragma unroll
    for (int r = 0; r < 8; ++r) dstv[r] = __shfl(dst_i, mrow + r, 32);

    __syncthreads();

    float macc[8];
    #pragma unroll
    for (int r = 0; r < 8; ++r) macc[r] = 0.f;

    int col = lane & 15;
    #pragma unroll
    for (int t = 0; t < 16; ++t) {
        v16h bm;
        #pragma unroll
        for (int i = 0; i < 16; ++i) bm[i] = (_Float16)0.f;
        if (lane < 16) {  // B: lanes 0-15 hold K=0..7 at column N=lane
            v8h wv = *reinterpret_cast<const v8h*>(&Wl[(t * 16 + lane) * DE]);
            #pragma unroll
            for (int k = 0; k < DE; ++k) bm[k] = wv[k];
        }
        float bias = benl[t * 16 + col];
        v8f c;
        #pragma unroll
        for (int r = 0; r < 8; ++r) c[r] = bias;
        v8f d = __builtin_amdgcn_wmma_f32_16x16x32_f16(false, a, false, bm,
                                                       (short)0, c, false, false);
        #pragma unroll
        for (int r = 0; r < 8; ++r) {
            float w = fast_tanh(d[r]);
            macc[r] = fmaf(hs[wave][(mrow + r) * 17 + t], w, macc[r]);
        }
    }
    #pragma unroll
    for (int r = 0; r < 8; ++r)
        atomicAdd(&h_out[(size_t)dstv[r] * DD + col], macc[r]);
}

// ---------------- masked segment mean/max pooling ----------------------------
__global__ void k_pool(const float* __restrict__ h2, const unsigned char* __restrict__ mask,
                       const int* __restrict__ batch, float* __restrict__ psum,
                       float* __restrict__ pcnt, unsigned* __restrict__ pmax) {
    int idx = blockIdx.x * blockDim.x + threadIdx.x;
    if (idx >= NN * DD) return;
    int n = idx >> 4, c = idx & 15;
    if (!mask[n]) return;
    int g = batch[n];
    float v = h2[idx];
    atomicAdd(&psum[g * DD + c], v);
    atomicMax(&pmax[g * DD + c], fkey(v));
    if (c == 0) atomicAdd(&pcnt[g], 1.f);
}

// ---------------- predictor head ---------------------------------------------
__global__ void k_head(const float* __restrict__ psum, const float* __restrict__ pcnt,
                       const unsigned* __restrict__ pmax, const float* __restrict__ W1,
                       const float* __restrict__ b1, const float* __restrict__ W2,
                       const float* __restrict__ b2, float* __restrict__ out) {
    int g = threadIdx.x;
    if (g >= NG) return;
    float cnt = fmaxf(pcnt[g], 1.f);
    float pooled[2 * DD];
    #pragma unroll
    for (int c = 0; c < DD; ++c) {
        pooled[c] = psum[g * DD + c] / cnt;
        pooled[DD + c] = fkey_inv(pmax[g * DD + c]);
    }
    float t1[5];
    #pragma unroll
    for (int j = 0; j < 5; ++j) {
        float s = b1[j];
        #pragma unroll
        for (int k = 0; k < 2 * DD; ++k) s = fmaf(pooled[k], W1[k * 5 + j], s);
        t1[j] = fmaxf(s, 0.f);
    }
    float s2 = b2[0];
    #pragma unroll
    for (int j = 0; j < 5; ++j) s2 = fmaf(t1[j], W2[j], s2);
    out[g] = fmaxf(s2, 0.f);
}

extern "C" void kernel_launch(void* const* d_in, const int* in_sizes, int n_in,
                              void* d_out, int out_size, void* d_ws, size_t ws_size,
                              hipStream_t stream) {
    const float* x         = (const float*)d_in[0];
    const float* edge_attr = (const float*)d_in[1];
    const int*   eidx      = (const int*)d_in[2];
    const unsigned char* mask = (const unsigned char*)d_in[3];
    const int*   batch     = (const int*)d_in[4];
    const float* W_ne = (const float*)d_in[5],  *b_ne = (const float*)d_in[6];
    const float* g_nbn = (const float*)d_in[7], *be_nbn = (const float*)d_in[8];
    const float* W_ee = (const float*)d_in[9],  *b_ee = (const float*)d_in[10];
    const float* g_ebn = (const float*)d_in[11], *be_ebn = (const float*)d_in[12];
    const float* W_en1 = (const float*)d_in[13], *b_en1 = (const float*)d_in[14];
    const float* root1 = (const float*)d_in[15], *bias1 = (const float*)d_in[16];
    const float* W_en2 = (const float*)d_in[17], *b_en2 = (const float*)d_in[18];
    const float* root2 = (const float*)d_in[19], *bias2 = (const float*)d_in[20];
    const float* W_p1 = (const float*)d_in[21],  *b_p1 = (const float*)d_in[22];
    const float* W_p2 = (const float*)d_in[23],  *b_p2 = (const float*)d_in[24];
    float* out = (float*)d_out;

    float* ws    = (float*)d_ws;
    float* h0    = ws;                 // N*16
    float* h1    = h0 + (size_t)NN * DD;
    float* h2    = h1 + (size_t)NN * DD;
    float* stats = h2 + (size_t)NN * DD;   // 96 floats
    float* psum  = stats + 96;             // G*16
    float* pcnt  = psum + NG * DD;         // G
    unsigned* pmax = (unsigned*)(pcnt + NG);  // G*16

    const int TB = 256;
    const int nblk = (NN + TB - 1) / TB;        // 196
    const int eblk = (NE + TB - 1) / TB;        // 3125
    const int cblk = NE / (16 * 8);             // 6250, exact
    const int pblk = (NN * DD + TB - 1) / TB;   // 3125

    k_init<<<4, TB, 0, stream>>>(stats, psum, pcnt, pmax);
    k_node_stats<<<nblk, TB, 0, stream>>>(x, W_ne, b_ne, stats, stats + 16);
    k_edge_stats<<<eblk, TB, 0, stream>>>(edge_attr, W_ee, b_ee, stats + 64, stats + 72);
    k_finalize<<<1, 32, 0, stream>>>(g_nbn, be_nbn, g_ebn, be_ebn, stats);
    k_node_apply<<<nblk, TB, 0, stream>>>(x, W_ne, b_ne, stats, h0);
    k_root<<<nblk, TB, 0, stream>>>(h0, h1, root1, bias1);
    k_nnconv<<<cblk, TB, 0, stream>>>(h0, h1, edge_attr, eidx, W_ee, b_ee, W_en1, b_en1, stats);
    k_root<<<nblk, TB, 0, stream>>>(h1, h2, root2, bias2);
    k_nnconv<<<cblk, TB, 0, stream>>>(h1, h2, edge_attr, eidx, W_ee, b_ee, W_en2, b_en2, stats);
    k_pool<<<pblk, TB, 0, stream>>>(h2, mask, batch, psum, pcnt, pmax);
    k_head<<<1, 64, 0, stream>>>(psum, pcnt, pmax, W_p1, b_p1, W_p2, b_p2, out);
    (void)in_sizes; (void)n_in; (void)out_size; (void)ws_size;
}
